// WindowMultiheadSelfAttention_28991029248670
// MI455X (gfx1250) — matrix operand order repaired and checked by reading, not compiled, not run on hardware
//
#include <hip/hip_runtime.h>
#include <hip/hip_bf16.h>

// ---------------------------------------------------------------------------
// CDNA5 / gfx1250 fused Swin window attention.
// bf16 WMMA (v_wmma_f32_16x16x32_bf16) for all GEMMs, wave32, WGP-resident
// per-window attention. Window partition + cyclic shift fused into the
// projection gathers/scatters. Q/K window staging via the Tensor Data Mover
// (tensor_load_to_lds + s_wait_tensorcnt) with hardware LDS row padding.
// ---------------------------------------------------------------------------

typedef __attribute__((ext_vector_type(16))) __bf16 v16bf;
typedef __attribute__((ext_vector_type(8)))  float  v8f;
typedef unsigned int u32x4 __attribute__((ext_vector_type(4)));
typedef int          i32x8 __attribute__((ext_vector_type(8)));
typedef int          i32x4 __attribute__((ext_vector_type(4)));

#if defined(__has_builtin)
#if __has_builtin(__builtin_amdgcn_tensor_load_to_lds)
#define HAVE_TDM 1
#endif
#endif
#ifndef HAVE_TDM
#define HAVE_TDM 0
#endif

#if defined(__has_include)
#if __has_include(<hip/amd_detail/amd_gfx1250_TDM.h>)
#define TDM_6ARG 1
#endif
#endif
#ifndef TDM_6ARG
#define TDM_6ARG 0
#endif

// ---- helpers ---------------------------------------------------------------

__device__ __forceinline__ __bf16 us2bf(unsigned short u) {
    union { unsigned short u; __bf16 b; } c; c.u = u; return c.b;
}

__device__ __forceinline__ unsigned short f2bf(float f) {
    unsigned int x = __float_as_uint(f);
    unsigned int r = (x + 0x7fffu + ((x >> 16) & 1u)) >> 16;   // RNE
    return (unsigned short)r;
}

__device__ __forceinline__ v8f wmma_bf16(v16bf a, v16bf b, v8f c) {
    // D = A(16x32 bf16) x B(32x16 bf16) + C(16x16 f32)
    return __builtin_amdgcn_wmma_f32_16x16x32_bf16(
        /*neg_a=*/false, a, /*neg_b=*/false, b,
        /*c_mod=*/(short)0, c, /*reuse_a=*/false, /*reuse_b=*/false);
}

// 16-bit A-matrix 16x32 fragment (this file's contiguous K order, matching load_b_frag):
// lane l<16: row=l,   K = {0..15}; lane l>=16: row=l-16, K = {16..31}
__device__ __forceinline__ v16bf load_a_frag(const unsigned short* p, int stride,
                                             int row0, int col0) {
    const int l    = threadIdx.x & 31;
    const int k0q  = (l >> 4) * 16;
    const unsigned short* q = p + (row0 + (l & 15)) * stride + col0 + k0q;
    v16bf a;
#pragma unroll
    for (int e = 0; e < 8; ++e) { a[e] = us2bf(q[e]); a[8 + e] = us2bf(q[8 + e]); }
    return a;
}

// 16-bit B-matrix 32x16 fragment, source stored "N-major" (row = N, col = K):
// lane l: N column = l&15, K = 16*(l>>4) + e, e=0..15
__device__ __forceinline__ v16bf load_b_frag(const unsigned short* p, int stride,
                                             int row0, int col0) {
    const int l  = threadIdx.x & 31;
    const int k0 = (l >> 4) * 16;
    const unsigned short* q = p + (row0 + (l & 15)) * stride + col0 + k0;
    v16bf b;
#pragma unroll
    for (int e = 0; e < 16; ++e) b[e] = us2bf(q[e]);
    return b;
}

#define WS_ 7
#define NH_ 8
#define DM_ 256
#define IMG_ 56
#define S_ 49
#define ROWS_ (32 * 3136)     // B*N = windowed rows total

// ---------------------------------------------------------------------------
// Kernel 1: QKV projection. y = x @ w^T + b, rows gathered in window order
// with the -4 cyclic shift applied. Output bf16, windowed layout.
// grid = (1568, 6): 64-row tiles x {q,k,v}x{lo128,hi128}
// ---------------------------------------------------------------------------
__global__ __launch_bounds__(256) void qkv_proj_kernel(
    const float* __restrict__ x,
    const float* __restrict__ q_w, const float* __restrict__ q_b,
    const float* __restrict__ k_w, const float* __restrict__ k_b,
    const float* __restrict__ v_w, const float* __restrict__ v_b,
    const int*   __restrict__ shift,
    unsigned short* __restrict__ Qo, unsigned short* __restrict__ Ko,
    unsigned short* __restrict__ Vo)
{
    __shared__ unsigned short As[64 * 40];    // 64 rows x 32 K (pad 40)
    __shared__ unsigned short Bs[128 * 40];   // 128 N x 32 K (N-major)
    __shared__ int srcRow[64];

    const int tid     = threadIdx.x;
    const int mBase   = blockIdx.x * 64;
    const int nb      = blockIdx.y;           // 0..5
    const int proj    = nb >> 1;              // 0=q 1=k 2=v
    const int colBase = (nb & 1) * 128;
    const float* w    = proj == 0 ? q_w : (proj == 1 ? k_w : v_w);
    const float* bias = proj == 0 ? q_b : (proj == 1 ? k_b : v_b);
    unsigned short* outp = proj == 0 ? Qo : (proj == 1 ? Ko : Vo);
    const int do_shift = (*shift != 0);

    if (tid < 64) {
        int r  = mBase + tid;
        int b  = r / 3136, rr = r % 3136;
        int win = rr / S_, pos = rr % S_;
        int wh = win >> 3, ww = win & 7;
        int ii = pos / WS_, jj = pos % WS_;
        int gi = wh * WS_ + ii, gj = ww * WS_ + jj;
        if (do_shift) { gi = (gi + 4) % IMG_; gj = (gj + 4) % IMG_; } // roll(-4)
        srcRow[tid] = b * 3136 + gi * IMG_ + gj;
    }
    __syncthreads();

    const int wave = tid >> 5, wm = wave >> 1, wn = wave & 1;
    const int l = tid & 31, half = l >> 4, ln = l & 15;

    v8f acc[4];
    const v8f vzero = {0.f, 0.f, 0.f, 0.f, 0.f, 0.f, 0.f, 0.f};
#pragma unroll
    for (int j = 0; j < 4; ++j) acc[j] = vzero;

    for (int kk = 0; kk < DM_; kk += 32) {
        {   // stage A tile (gathered rows, f32 -> bf16)
            int row = tid >> 2, c0 = (tid & 3) * 8;
            const float* src = x + (size_t)srcRow[row] * DM_ + kk + c0;
            if (kk + 32 < DM_) __builtin_prefetch(src + 32, 0, 1);  // next chunk
            unsigned short* dst = &As[row * 40 + c0];
#pragma unroll
            for (int u = 0; u < 8; ++u) dst[u] = f2bf(src[u]);
        }
        {   // stage B tile: Bs[n][k] = w[colBase+n][kk+k]
            int n = tid >> 1, c0 = (tid & 1) * 16;
            const float* src = w + (size_t)(colBase + n) * DM_ + kk + c0;
            if (kk + 32 < DM_) __builtin_prefetch(src + 32, 0, 1);
            unsigned short* dst = &Bs[n * 40 + c0];
#pragma unroll
            for (int u = 0; u < 16; ++u) dst[u] = f2bf(src[u]);
        }
        __syncthreads();
        v16bf a = load_a_frag(As, 40, wm * 16, 0);
#pragma unroll
        for (int j = 0; j < 4; ++j) {
            v16bf b = load_b_frag(Bs, 40, wn * 64 + j * 16, 0);
            acc[j] = wmma_bf16(a, b, acc[j]);
        }
        __syncthreads();
    }

#pragma unroll
    for (int j = 0; j < 4; ++j) {
        int col = colBase + wn * 64 + j * 16 + ln;
        float bb = bias[col];
#pragma unroll
        for (int v = 0; v < 8; ++v) {
            int row = mBase + wm * 16 + v + half * 8;
            outp[(size_t)row * DM_ + col] = f2bf(acc[j][v] + bb);
        }
    }
}

// ---------------------------------------------------------------------------
// Kernel 2: per-window attention. One block per window, one wave per head.
// S = QK^T/sqrt(dk) + bias + shift-mask, softmax in registers, O = P V.
// 49 padded to 64 -> 4x4 and 4x2 tiles of 16x16x32 bf16 WMMA.
// Q/K staged by the Tensor Data Mover with HW row padding when available.
// ---------------------------------------------------------------------------
#define QS_STRIDE 264   // 256 + 8 pad (ushorts) == 512B + 16B pad per row
#define VT_STRIDE 72    // 64 + 8 pad
#define P_STRIDE  72

#if HAVE_TDM
// Issue one TDM descriptor: 2D tile 49 rows x 128 DWORDs (bf16 512B/row),
// LDS destination padded by 4 DWORDs after every 128 DWORDs (-> 528B stride).
__device__ __forceinline__ void tdm_load_qk(const unsigned short* gsrc,
                                            unsigned lds_byte_addr) {
    unsigned long long ga = (unsigned long long)(size_t)gsrc;
    u32x4 g0;
    g0[0] = 1u;                                  // count=1 (valid user D#)
    g0[1] = lds_byte_addr;                       // lds_addr
    g0[2] = (unsigned)(ga & 0xffffffffu);        // global_addr[31:0]
    g0[3] = (unsigned)((ga >> 32) & 0x01ffffffu) // global_addr[56:32]
            | (2u << 30);                        // type=2 ("image")
    i32x8 g1;
    g1[0] = (2 << 16)      // data_size = 4 bytes
          | (1 << 20)      // pad_enable
          | (6 << 22)      // pad_interval: 128 DWORDs
          | (3 << 25);     // pad_amount: 4 DWORDs (16B)
    g1[1] = (128 & 0xffff) << 16;        // tensor_dim0[15:0] = 128 DWORDs/row
    g1[2] = (S_ & 0xffff) << 16;         // tensor_dim0 hi = 0, tensor_dim1 = 49
    g1[3] = (128 << 16);                 // tensor_dim1 hi = 0, tile_dim0 = 128
    g1[4] = S_;                          // tile_dim1 = 49, tile_dim2 = 0
    g1[5] = 128;                         // tensor_dim0_stride = 128 DWORDs
    g1[6] = 0;                           // stride0 hi, dim1_stride lo
    g1[7] = 0;
    i32x4 z4 = {0, 0, 0, 0};
#if TDM_6ARG
    i32x8 z8 = {0, 0, 0, 0, 0, 0, 0, 0};
    __builtin_amdgcn_tensor_load_to_lds(g0, g1, z4, z4, z8, 0);
#else
    __builtin_amdgcn_tensor_load_to_lds(g0, g1, z4, z4, 0);
#endif
}
#endif

__global__ __launch_bounds__(256) void swin_attn_kernel(
    const unsigned short* __restrict__ Q,
    const unsigned short* __restrict__ K,
    const unsigned short* __restrict__ V,
    const float* __restrict__ bias_table,
    const int*   __restrict__ rel_index,
    const int*   __restrict__ shift,
    unsigned short* __restrict__ O)
{
    extern __shared__ unsigned short smem[];
    unsigned short* Qs = smem;                          // 64 x QS_STRIDE
    unsigned short* Ks = Qs + 64 * QS_STRIDE;           // 64 x QS_STRIDE
    unsigned short* Vt = Ks + 64 * QS_STRIDE;           // 256 feat x VT_STRIDE
    unsigned short* P  = Vt + 256 * VT_STRIDE;          // 8 heads x 64 x P_STRIDE

    const int tid = threadIdx.x;
    const int win = blockIdx.x;                 // b*64 + wh*8 + ww
    const int wh  = (win >> 3) & 7;
    const int ww  = win & 7;
    const size_t rowBase = (size_t)win * S_;
    const int do_shift = (*shift != 0);

#if HAVE_TDM
    // ---- DMA rows 0..48 of Q and K into padded LDS via the TDM ----
    if (tid < 32) {
        tdm_load_qk(Q + rowBase * DM_, (unsigned)(size_t)(void*)Qs);
    } else if (tid < 64) {
        tdm_load_qk(K + rowBase * DM_, (unsigned)(size_t)(void*)Ks);
    }
    // zero pad rows 49..63 (disjoint from TDM destination rows)
    for (int idx = tid; idx < 15 * 132; idx += 256) {
        int r = S_ + idx / 132, cu = idx % 132;
        ((unsigned int*)(Qs + r * QS_STRIDE))[cu] = 0u;
        ((unsigned int*)(Ks + r * QS_STRIDE))[cu] = 0u;
    }
#else
    // ---- manual stage Q, K (uint copies, zero-pad rows 49..63) ----
    for (int idx = tid; idx < 64 * 128; idx += 256) {
        int r = idx >> 7, cu = idx & 127;
        unsigned int qv = 0u, kv = 0u;
        if (r < S_) {
            qv = ((const unsigned int*)(Q + (rowBase + r) * DM_))[cu];
            kv = ((const unsigned int*)(K + (rowBase + r) * DM_))[cu];
        }
        ((unsigned int*)(Qs + r * QS_STRIDE))[cu] = qv;
        ((unsigned int*)(Ks + r * QS_STRIDE))[cu] = kv;
    }
#endif
    // ---- stage V transposed: Vt[feature][row] (overlaps with the TDM DMA) ----
    for (int idx = tid; idx < 64 * 256; idx += 256) {
        int r = idx >> 8, c = idx & 255;
        unsigned short vv = (r < S_) ? V[(rowBase + r) * DM_ + c] : (unsigned short)0;
        Vt[c * VT_STRIDE + r] = vv;
    }
#if HAVE_TDM
    __builtin_amdgcn_s_wait_tensorcnt(0);       // issuing waves drain TENSORcnt
#endif
    __syncthreads();

    const int h = tid >> 5;                     // head == wave
    const int l = tid & 31, half = l >> 4, ln = l & 15;

    const v8f vzero = {0.f, 0.f, 0.f, 0.f, 0.f, 0.f, 0.f, 0.f};
    v8f acc[4][4];
#pragma unroll
    for (int i = 0; i < 4; ++i)
#pragma unroll
        for (int j = 0; j < 4; ++j) acc[i][j] = vzero;

    // ---- S = Q K^T (K^T fragment = row-major K rows as B N-columns) ----
#pragma unroll
    for (int i = 0; i < 4; ++i) {
        v16bf a = load_a_frag(Qs, QS_STRIDE, i * 16, h * 32);
#pragma unroll
        for (int j = 0; j < 4; ++j) {
            v16bf b = load_b_frag(Ks, QS_STRIDE, j * 16, h * 32);
            acc[i][j] = wmma_bf16(a, b, acc[i][j]);
        }
    }

    // ---- bias + shift mask + softmax (rows across 16 lanes, shfl reduce) ----
    const float scale = 0.17677669529663687f;   // 1/sqrt(32)
    unsigned short* Ph = P + h * 64 * P_STRIDE;
#pragma unroll
    for (int i = 0; i < 4; ++i) {
#pragma unroll
        for (int v = 0; v < 8; ++v) {
            const int q = i * 16 + v + half * 8;
            float s[4];
#pragma unroll
            for (int j = 0; j < 4; ++j) {
                int k = j * 16 + ln;
                float val = acc[i][j][v] * scale;
                if (q < S_ && k < S_) {
                    int ridx = rel_index[q * S_ + k];
                    val += bias_table[ridx * NH_ + h];
                    if (do_shift) {
                        int gqi = wh * WS_ + q / WS_, gqj = ww * WS_ + q % WS_;
                        int gki = wh * WS_ + k / WS_, gkj = ww * WS_ + k % WS_;
                        int rq = (gqi < 49 ? 0 : (gqi < 53 ? 1 : 2)) * 3 +
                                 (gqj < 49 ? 0 : (gqj < 53 ? 1 : 2));
                        int rk = (gki < 49 ? 0 : (gki < 53 ? 1 : 2)) * 3 +
                                 (gkj < 49 ? 0 : (gkj < 53 ? 1 : 2));
                        if (rq != rk) val += -100.0f;
                    }
                } else {
                    val = -1e30f;               // pad columns excluded
                }
                s[j] = val;
            }
            float m = fmaxf(fmaxf(s[0], s[1]), fmaxf(s[2], s[3]));
#pragma unroll
            for (int d = 1; d < 16; d <<= 1) m = fmaxf(m, __shfl_xor(m, d, 32));
            float e0 = __expf(s[0] - m), e1 = __expf(s[1] - m);
            float e2 = __expf(s[2] - m), e3 = __expf(s[3] - m);
            float sum = e0 + e1 + e2 + e3;
#pragma unroll
            for (int d = 1; d < 16; d <<= 1) sum += __shfl_xor(sum, d, 32);
            float inv = 1.0f / sum;
            Ph[q * P_STRIDE + 0 * 16 + ln] = f2bf(e0 * inv);
            Ph[q * P_STRIDE + 1 * 16 + ln] = f2bf(e1 * inv);
            Ph[q * P_STRIDE + 2 * 16 + ln] = f2bf(e2 * inv);
            Ph[q * P_STRIDE + 3 * 16 + ln] = f2bf(e3 * inv);
        }
    }
    __syncthreads();

    // ---- O = P V ----
    v8f o[4][2];
#pragma unroll
    for (int i = 0; i < 4; ++i) { o[i][0] = vzero; o[i][1] = vzero; }
#pragma unroll
    for (int kk = 0; kk < 2; ++kk) {
#pragma unroll
        for (int i = 0; i < 4; ++i) {
            v16bf a = load_a_frag(Ph, P_STRIDE, i * 16, kk * 32);
#pragma unroll
            for (int jj = 0; jj < 2; ++jj) {
                v16bf b = load_b_frag(Vt, VT_STRIDE, h * 32 + jj * 16, kk * 32);
                o[i][jj] = wmma_bf16(a, b, o[i][jj]);
            }
        }
    }
#pragma unroll
    for (int i = 0; i < 4; ++i)
#pragma unroll
        for (int jj = 0; jj < 2; ++jj)
#pragma unroll
            for (int v = 0; v < 8; ++v) {
                int q = i * 16 + v + half * 8;
                if (q < S_) {
                    int col = h * 32 + jj * 16 + ln;
                    O[(rowBase + q) * DM_ + col] = f2bf(o[i][jj][v]);
                }
            }
}

// ---------------------------------------------------------------------------
// Kernel 3: output projection + un-window + roll(+3) scatter, fp32 out.
// grid = (1568, 2)
// ---------------------------------------------------------------------------
__global__ __launch_bounds__(256) void out_proj_kernel(
    const unsigned short* __restrict__ A,     // windowed attn out, bf16
    const float* __restrict__ o_w, const float* __restrict__ o_b,
    const int*   __restrict__ shift,
    float* __restrict__ out)
{
    __shared__ unsigned short As[64 * 40];
    __shared__ unsigned short Bs[128 * 40];
    __shared__ int dstRow[64];

    const int tid   = threadIdx.x;
    const int mBase = blockIdx.x * 64;
    const int colBase = blockIdx.y * 128;
    const int do_shift = (*shift != 0);

    if (tid < 64) {
        int r  = mBase + tid;
        int b  = r / 3136, rr = r % 3136;
        int win = rr / S_, pos = rr % S_;
        int wh = win >> 3, ww = win & 7;
        int ii = pos / WS_, jj = pos % WS_;
        int gi = wh * WS_ + ii, gj = ww * WS_ + jj;
        if (do_shift) { gi = (gi + 3) % IMG_; gj = (gj + 3) % IMG_; } // roll(+3)
        dstRow[tid] = b * 3136 + gi * IMG_ + gj;
    }
    __syncthreads();

    const int wave = tid >> 5, wm = wave >> 1, wn = wave & 1;
    const int l = tid & 31, half = l >> 4, ln = l & 15;

    v8f acc[4];
    const v8f vzero = {0.f, 0.f, 0.f, 0.f, 0.f, 0.f, 0.f, 0.f};
#pragma unroll
    for (int j = 0; j < 4; ++j) acc[j] = vzero;

    for (int kk = 0; kk < DM_; kk += 32) {
        {
            int row = tid >> 2, c0 = (tid & 3) * 8;
            const unsigned short* src = A + (size_t)(mBase + row) * DM_ + kk + c0;
            if (kk + 32 < DM_) __builtin_prefetch(src + 32, 0, 1);
            unsigned short* dst = &As[row * 40 + c0];
#pragma unroll
            for (int u = 0; u < 8; ++u) dst[u] = src[u];
        }
        {
            int n = tid >> 1, c0 = (tid & 1) * 16;
            const float* src = o_w + (size_t)(colBase + n) * DM_ + kk + c0;
            if (kk + 32 < DM_) __builtin_prefetch(src + 32, 0, 1);
            unsigned short* dst = &Bs[n * 40 + c0];
#pragma unroll
            for (int u = 0; u < 16; ++u) dst[u] = f2bf(src[u]);
        }
        __syncthreads();
        v16bf a = load_a_frag(As, 40, wm * 16, 0);
#pragma unroll
        for (int j = 0; j < 4; ++j) {
            v16bf b = load_b_frag(Bs, 40, wn * 64 + j * 16, 0);
            acc[j] = wmma_bf16(a, b, acc[j]);
        }
        __syncthreads();
    }

#pragma unroll
    for (int j = 0; j < 4; ++j) {
        int col = colBase + wn * 64 + j * 16 + ln;
        float bb = o_b[col];
#pragma unroll
        for (int v = 0; v < 8; ++v) {
            int rowL = wm * 16 + v + half * 8;
            out[(size_t)dstRow[rowL] * DM_ + col] = acc[j][v] + bb;
        }
    }
}

// ---------------------------------------------------------------------------

extern "C" void kernel_launch(void* const* d_in, const int* in_sizes, int n_in,
                              void* d_out, int out_size, void* d_ws, size_t ws_size,
                              hipStream_t stream) {
    (void)in_sizes; (void)n_in; (void)out_size; (void)ws_size;

    const float* x    = (const float*)d_in[0];
    const float* q_w  = (const float*)d_in[1];
    const float* q_b  = (const float*)d_in[2];
    const float* k_w  = (const float*)d_in[3];
    const float* k_b  = (const float*)d_in[4];
    const float* v_w  = (const float*)d_in[5];
    const float* v_b  = (const float*)d_in[6];
    const float* o_w  = (const float*)d_in[7];
    const float* o_b  = (const float*)d_in[8];
    const float* bias_table = (const float*)d_in[9];
    const int*   rel_index  = (const int*)d_in[10];
    const int*   shift      = (const int*)d_in[11];
    float* out = (float*)d_out;

    // scratch layout: Q | K | V | attn, all bf16 (ushort), windowed row order
    unsigned short* Qw = (unsigned short*)d_ws;
    unsigned short* Kw = Qw + (size_t)ROWS_ * DM_;
    unsigned short* Vw = Kw + (size_t)ROWS_ * DM_;
    unsigned short* Aw = Vw + (size_t)ROWS_ * DM_;

    dim3 gA(ROWS_ / 64, 6);
    qkv_proj_kernel<<<gA, 256, 0, stream>>>(x, q_w, q_b, k_w, k_b, v_w, v_b,
                                            shift, Qw, Kw, Vw);

    size_t smem = (size_t)(64 * QS_STRIDE + 64 * QS_STRIDE +
                           256 * VT_STRIDE + NH_ * 64 * P_STRIDE) *
                  sizeof(unsigned short);
    (void)hipFuncSetAttribute((const void*)swin_attn_kernel,
                              hipFuncAttributeMaxDynamicSharedMemorySize,
                              (int)smem);
    swin_attn_kernel<<<2048, 256, smem, stream>>>(Qw, Kw, Vw, bias_table,
                                                  rel_index, shift, Aw);

    dim3 gC(ROWS_ / 64, 2);
    out_proj_kernel<<<gC, 256, 0, stream>>>(Aw, o_w, o_b, shift, out);
}
